// DictLearn_10179072492160
// MI455X (gfx1250) — matrix operand order repaired and checked by reading, not compiled, not run on hardware
//
#include <hip/hip_runtime.h>
#include <hip/hip_bf16.h>
#include <math.h>

typedef __bf16 bf16_t;
typedef __bf16 v8bf  __attribute__((ext_vector_type(8)));
typedef __bf16 v16bf __attribute__((ext_vector_type(16)));
typedef float  v8f   __attribute__((ext_vector_type(8)));
typedef unsigned int v4u __attribute__((ext_vector_type(4)));
typedef int v8i __attribute__((ext_vector_type(8)));
typedef int v4i __attribute__((ext_vector_type(4)));

#define B_ROWS   8192
#define N_DIM    784
#define M_DIM    2048
#define LAMBDA_F 0.1f
#define N_ITER   50
#define P_ITER   100

#define TILE_K  32
#define TILE_MN 128
// LDS row: 32 bf16 = 64B data + 16B TDM pad = 80B  -> 40 halves, conflict-free & 16B aligned
#define LROW_H  40
#define LTILE_B (TILE_MN * LROW_H * 2)   // 10240 bytes per tile

// ---------------------------------------------------------------------------
// Tensor Data Mover: 2D tile load Global -> LDS (6-arg builtin form).
// rem0/rem1 = remaining tensor extent from tile start (TDM zero-fills OOB).
// pad: 4 DWORDs after every 16 DWORDs -> 80B LDS row stride.
// ---------------------------------------------------------------------------
__device__ __forceinline__ void tdm_load_2d(unsigned lds_off, const void* gsrc,
                                            unsigned rem0, unsigned rem1,
                                            unsigned tile0, unsigned tile1,
                                            unsigned long long stride0_elems) {
  unsigned long long ga = (unsigned long long)gsrc;
  v4u g0;
  g0[0] = 1u;                                              // count=1, user D#
  g0[1] = lds_off;                                         // lds_addr (bytes)
  g0[2] = (unsigned)ga;                                    // global_addr[31:0]
  g0[3] = (unsigned)((ga >> 32) & 0x1FFFFFFu) | (2u << 30);// addr[56:32] | type=2
  v8i g1;
  g1[0] = (int)((1u << 16) |      // data_size = 1 (2 bytes)
                (1u << 20) |      // pad_enable
                (3u << 22) |      // pad_interval: 16 DWORDs
                (3u << 25));      // pad_amount: 4 DWORDs
  g1[1] = (int)((rem0 & 0xFFFFu) << 16);                       // tensor_dim0 lo
  g1[2] = (int)((rem0 >> 16) | ((rem1 & 0xFFFFu) << 16));      // dim0 hi | dim1 lo
  g1[3] = (int)((rem1 >> 16) | (tile0 << 16));                 // dim1 hi | tile_dim0
  g1[4] = (int)(tile1 & 0xFFFFu);                              // tile_dim1, tile_dim2=0
  g1[5] = (int)(unsigned)(stride0_elems & 0xFFFFFFFFull);      // dim0_stride lo
  g1[6] = (int)(unsigned)((stride0_elems >> 32) & 0xFFFFull);  // dim0_stride hi
  g1[7] = 0;
  v4i zz = {0, 0, 0, 0};
  v8i zg = {0, 0, 0, 0, 0, 0, 0, 0};
  __builtin_amdgcn_tensor_load_to_lds(g0, g1, zz, zz, zg, 0);
}

__device__ __forceinline__ float softt(float x, float t) {
  float ax = fabsf(x) - t;
  ax = ax > 0.f ? ax : 0.f;
  return x < 0.f ? -ax : ax;
}

// ---------------------------------------------------------------------------
// Fused bf16 WMMA GEMM: C = A[MxK] * Bt[NxK]^T, fp32 accumulate.
// mode 0: residual  = acc - Y     -> outB (bf16), ||r||^2 atomically -> normSlot
// mode 1: final X   = acc         -> outF (fp32, d_out)
// mode 2: Gamma0    = soft(eta*acc, lambda)          -> gammaF, Zf, outB(=Zb)
// mode 3: step      = soft(Zf - eta*acc, thr); Znew  -> gammaF, Zf, outB(=Zb)
// mode 4: last step = like 3 but outB = bf16(Gamma)  (feeds final X GEMM)
// ---------------------------------------------------------------------------
__global__ __launch_bounds__(256) void wmma_gemm_fused(
    int mode, const bf16_t* __restrict__ A, const bf16_t* __restrict__ Bt,
    int M, int Nn, int K,
    const float* __restrict__ Yin, float* __restrict__ outF,
    bf16_t* __restrict__ outB, float* __restrict__ Zf,
    float* __restrict__ gammaF, const float* __restrict__ scal,
    float* __restrict__ normSlot, float mom) {
  __shared__ __align__(16) unsigned char lds[4 * LTILE_B];

  const int tid  = threadIdx.x;
  const int lane = tid & 31;
  const int wave = tid >> 5;
  const int wr   = wave & 3;   // *32 rows
  const int wc   = wave >> 2;  // *64 cols
  const int row0 = blockIdx.y * TILE_MN;
  const int col0 = blockIdx.x * TILE_MN;
  const int ksteps = (K + TILE_K - 1) / TILE_K;
  const unsigned ldsBase = (unsigned)(unsigned long long)(&lds[0]);

  v8f acc[2][4];
#pragma unroll
  for (int i = 0; i < 2; ++i)
#pragma unroll
    for (int j = 0; j < 4; ++j) {
      v8f z8 = {0.f, 0.f, 0.f, 0.f, 0.f, 0.f, 0.f, 0.f};
      acc[i][j] = z8;
    }

  if (tid == 0) {  // prologue TDM: A tile + B tile into buffer 0 (one wave issues)
    tdm_load_2d(ldsBase, A + (long long)row0 * K, (unsigned)K,
                (unsigned)(M - row0), TILE_K, TILE_MN, (unsigned long long)K);
    tdm_load_2d(ldsBase + 2 * LTILE_B, Bt + (long long)col0 * K, (unsigned)K,
                (unsigned)(Nn - col0), TILE_K, TILE_MN, (unsigned long long)K);
  }

  const int c0 = (lane >> 4) * 8;   // A chunks at halves c0 and c0+16
  const int kb = (lane >> 4) * 8;   // B chunks at halves kb and kb+16

  for (int kt = 0; kt < ksteps; ++kt) {
    const int s = kt & 1;
    if (tid == 0) __builtin_amdgcn_s_wait_tensorcnt(0);
    __syncthreads();  // tile s ready; all waves done with tile s^1
    if (tid == 0 && kt + 1 < ksteps) {
      const long long k0 = (long long)(kt + 1) * TILE_K;
      tdm_load_2d(ldsBase + (s ^ 1) * LTILE_B, A + (long long)row0 * K + k0,
                  (unsigned)(K - k0), (unsigned)(M - row0), TILE_K, TILE_MN,
                  (unsigned long long)K);
      tdm_load_2d(ldsBase + (2 + (s ^ 1)) * LTILE_B,
                  Bt + (long long)col0 * K + k0, (unsigned)(K - k0),
                  (unsigned)(Nn - col0), TILE_K, TILE_MN, (unsigned long long)K);
    }
    const bf16_t* At = (const bf16_t*)(lds + s * LTILE_B);
    const bf16_t* Bs = (const bf16_t*)(lds + (2 + s) * LTILE_B);

    v16bf afrag[2], bfrag[4];
#pragma unroll
    for (int i = 0; i < 2; ++i) {
      const int r = wr * 32 + i * 16 + (lane & 15);
      v8bf lo = *(const v8bf*)(At + r * LROW_H + c0);
      v8bf hi = *(const v8bf*)(At + r * LROW_H + c0 + 16);
#pragma unroll
      for (int t = 0; t < 8; ++t) { afrag[i][t] = lo[t]; afrag[i][t + 8] = hi[t]; }
    }
#pragma unroll
    for (int j = 0; j < 4; ++j) {
      const int r = wc * 64 + j * 16 + (lane & 15);
      v8bf lo = *(const v8bf*)(Bs + r * LROW_H + kb);
      v8bf hi = *(const v8bf*)(Bs + r * LROW_H + kb + 16);
#pragma unroll
      for (int t = 0; t < 8; ++t) { bfrag[j][t] = lo[t]; bfrag[j][t + 8] = hi[t]; }
    }
#pragma unroll
    for (int i = 0; i < 2; ++i)
#pragma unroll
      for (int j = 0; j < 4; ++j)
        acc[i][j] = __builtin_amdgcn_wmma_f32_16x16x32_bf16(
            false, afrag[i], false, bfrag[j], (short)0, acc[i][j], false, false);
  }

  // ---------------- epilogue ----------------
  const float eta = scal[1];
  const float thr = scal[2];
  float ss = 0.f;
#pragma unroll
  for (int i = 0; i < 2; ++i)
#pragma unroll
    for (int j = 0; j < 4; ++j)
#pragma unroll
      for (int v = 0; v < 8; ++v) {
        const int m = row0 + wr * 32 + i * 16 + v + ((lane >> 4) * 8);
        const int n = col0 + wc * 64 + j * 16 + (lane & 15);
        if (n < Nn) {
          const long long idx = (long long)m * Nn + n;
          const float a = acc[i][j][v];
          if (mode == 0) {
            const float r = a - Yin[idx];
            outB[idx] = (bf16_t)r;
            ss += r * r;
          } else if (mode == 1) {
            outF[idx] = a;
          } else if (mode == 2) {
            const float g = softt(eta * a, LAMBDA_F);
            gammaF[idx] = g; Zf[idx] = g; outB[idx] = (bf16_t)g;
          } else {
            const float g  = softt(Zf[idx] - eta * a, thr);
            const float zn = g + mom * (g - gammaF[idx]);
            gammaF[idx] = g; Zf[idx] = zn;
            outB[idx] = (mode == 4) ? (bf16_t)g : (bf16_t)zn;
          }
        }
      }
  if (mode == 0) {
    for (int off = 16; off; off >>= 1) ss += __shfl_down(ss, off, 32);
    if (lane == 0) atomicAdd(normSlot, ss);
  }
}

// ---------------------------------------------------------------------------
// Small helper kernels
// ---------------------------------------------------------------------------
__global__ void init_kernel(float* scal, float* X) {
  const int i = blockIdx.x * blockDim.x + threadIdx.x;
  if (i < 64) scal[i] = 0.f;
  if (i < M_DIM) {
    unsigned h = (unsigned)(i + 1) * 2654435761u;
    h ^= h >> 13; h *= 1274126177u; h ^= h >> 16;
    X[i] = (float)(h & 0xFFFFFF) * (1.0f / 8388608.0f) - 1.0f;  // [-1,1)
  }
}

__global__ void convW_kernel(const float* __restrict__ W, bf16_t* __restrict__ Wb,
                             bf16_t* __restrict__ Wtb) {
  const int i = blockIdx.x * 256 + threadIdx.x;
  if (i < N_DIM * M_DIM) {
    const int r = i / M_DIM, c = i % M_DIM;
    const float w = W[i];
    Wb[i] = (bf16_t)w;
    Wtb[(long long)c * N_DIM + r] = (bf16_t)w;
  }
}

__global__ void convY_kernel(const float* __restrict__ Y, bf16_t* __restrict__ Yb,
                             float* __restrict__ yss) {
  const long long n = (long long)B_ROWS * N_DIM;
  float s = 0.f;
  for (long long i = (long long)blockIdx.x * 256 + threadIdx.x; i < n;
       i += (long long)gridDim.x * 256) {
    const float y = Y[i];
    Yb[i] = (bf16_t)y;
    s += y * y;
  }
  for (int off = 16; off; off >>= 1) s += __shfl_down(s, off, 32);
  if ((threadIdx.x & 31) == 0) atomicAdd(yss, s);
}

// power method: t = X @ W^T  (784 dots of length 2048)
__global__ void pm1_kernel(const float* __restrict__ W, const float* __restrict__ X,
                           float* __restrict__ tvec) {
  const int row = blockIdx.x * 32 + (threadIdx.x >> 3);
  const int l8 = threadIdx.x & 7;
  float s = 0.f;
  if (row < N_DIM)
    for (int k = l8; k < M_DIM; k += 8) s += X[k] * W[(long long)row * M_DIM + k];
  s += __shfl_down(s, 4, 8);
  s += __shfl_down(s, 2, 8);
  s += __shfl_down(s, 1, 8);
  if (row < N_DIM && l8 == 0) tvec[row] = s;
}

// power method: xnew = t @ W  (2048 dots of length 784, coalesced across lanes)
__global__ void pm2_kernel(const float* __restrict__ W, const float* __restrict__ tvec,
                           float* __restrict__ xnew) {
  const int m = blockIdx.x * 256 + threadIdx.x;
  float s = 0.f;
  for (int nr = 0; nr < N_DIM; ++nr) s += tvec[nr] * W[(long long)nr * M_DIM + m];
  xnew[m] = s;
}

__global__ void pm3_kernel(float* __restrict__ scal, const float* __restrict__ xnew,
                           float* __restrict__ X) {
  __shared__ float red[8];
  __shared__ float snrm;
  float s = 0.f;
  for (int i = threadIdx.x; i < M_DIM; i += 256) { const float v = xnew[i]; s += v * v; }
  for (int off = 16; off; off >>= 1) s += __shfl_down(s, off, 32);
  if ((threadIdx.x & 31) == 0) red[threadIdx.x >> 5] = s;
  __syncthreads();
  if (threadIdx.x == 0) {
    float tot = 0.f;
    for (int i = 0; i < 8; ++i) tot += red[i];
    const float nrm = sqrtf(tot);
    scal[0] = nrm;              // c
    scal[1] = 1.f / nrm;        // eta
    scal[2] = LAMBDA_F / nrm;   // lambda / c
    snrm = nrm;
  }
  __syncthreads();
  const float inv = 1.f / snrm;
  for (int i = threadIdx.x; i < M_DIM; i += 256) X[i] = xnew[i] * inv;
}

__global__ void fin_kernel(const float* __restrict__ scal, float* __restrict__ norms) {
  const int i = threadIdx.x;
  if (i < N_ITER) norms[i] = sqrtf(scal[4 + i] / scal[3]);
}

// ---------------------------------------------------------------------------
extern "C" void kernel_launch(void* const* d_in, const int* in_sizes, int n_in,
                              void* d_out, int out_size, void* d_ws, size_t ws_size,
                              hipStream_t stream) {
  const float* Y = (const float*)d_in[0];  // [8192, 784]
  const float* W = (const float*)d_in[1];  // [784, 2048]
  float* outX     = (float*)d_out;                                     // [8192,784]
  float* outGamma = outX + (long long)B_ROWS * N_DIM;                  // [8192,2048]
  float* outNorms = outGamma + (long long)B_ROWS * M_DIM;              // [50]

  char* ws = (char*)d_ws;
  size_t off = 0;
  auto alloc = [&](size_t bytes) -> void* {
    void* p = ws + off;
    off += (bytes + 255) & ~(size_t)255;
    return p;
  };
  float*  scal = (float*)alloc(64 * 4);              // c, eta, thr, yss, norm_acc[50]
  float*  Xv   = (float*)alloc(M_DIM * 4);
  float*  tvec = (float*)alloc(N_DIM * 4);
  float*  xnew = (float*)alloc(M_DIM * 4);
  bf16_t* Wb   = (bf16_t*)alloc((size_t)N_DIM * M_DIM * 2);  // W     bf16 [784,2048]
  bf16_t* Wtb  = (bf16_t*)alloc((size_t)N_DIM * M_DIM * 2);  // W^T   bf16 [2048,784]
  bf16_t* Rb   = (bf16_t*)alloc((size_t)B_ROWS * N_DIM * 2); // resid bf16 [8192,784]
  bf16_t* Zb   = (bf16_t*)alloc((size_t)B_ROWS * M_DIM * 2); // Z     bf16 [8192,2048]
  float*  Zf   = (float*)alloc((size_t)B_ROWS * M_DIM * 4);  // Z     f32

  // data-independent FISTA momentum sequence (host side)
  float mom[N_ITER];
  double t = 1.0;
  for (int i = 0; i < N_ITER; ++i) {
    const double tn = (1.0 + sqrt(1.0 + 4.0 * t * t)) / 2.0;
    mom[i] = (float)((t - 1.0) / tn);
    t = tn;
  }

  init_kernel<<<(M_DIM + 255) / 256, 256, 0, stream>>>(scal, Xv);
  convW_kernel<<<(N_DIM * M_DIM + 255) / 256, 256, 0, stream>>>(W, Wb, Wtb);
  convY_kernel<<<1024, 256, 0, stream>>>(Y, Rb, scal + 3);  // Rb <- bf16(Y), yss

  // power method: c = ||W||_2^2
  for (int p = 0; p < P_ITER; ++p) {
    pm1_kernel<<<(N_DIM + 31) / 32, 256, 0, stream>>>(W, Xv, tvec);
    pm2_kernel<<<M_DIM / 256, 256, 0, stream>>>(W, tvec, xnew);
    pm3_kernel<<<1, 256, 0, stream>>>(scal, xnew, Xv);
  }

  const dim3 blk(256);
  const dim3 gN(M_DIM / TILE_MN, B_ROWS / TILE_MN);                 // N=2048 GEMMs
  const dim3 gn((N_DIM + TILE_MN - 1) / TILE_MN, B_ROWS / TILE_MN); // N=784 GEMMs

  // Gamma0 = soft(eta * (Y @ W), lambda);  Z0 = Gamma0
  wmma_gemm_fused<<<gN, blk, 0, stream>>>(2, Rb, Wtb, B_ROWS, M_DIM, N_DIM,
                                          nullptr, nullptr, Zb, Zf, outGamma,
                                          scal, nullptr, 0.f);
  for (int i = 0; i < N_ITER; ++i) {
    // residual = Z @ W^T - Y   (Bt = W), ||r||^2 -> norm_acc[i]
    wmma_gemm_fused<<<gn, blk, 0, stream>>>(0, Zb, Wb, B_ROWS, N_DIM, M_DIM,
                                            Y, nullptr, Rb, nullptr, nullptr,
                                            scal, scal + 4 + i, 0.f);
    // Gamma/Z update: G = residual @ W   (Bt = W^T)
    wmma_gemm_fused<<<gN, blk, 0, stream>>>(i == N_ITER - 1 ? 4 : 3, Rb, Wtb,
                                            B_ROWS, M_DIM, N_DIM, nullptr,
                                            nullptr, Zb, Zf, outGamma, scal,
                                            nullptr, mom[i]);
  }
  // X = Gamma @ W^T  (Zb holds bf16(Gamma) after mode-4 epilogue)
  wmma_gemm_fused<<<gn, blk, 0, stream>>>(1, Zb, Wb, B_ROWS, N_DIM, M_DIM,
                                          nullptr, outX, nullptr, nullptr,
                                          nullptr, scal, nullptr, 0.f);
  fin_kernel<<<1, 64, 0, stream>>>(scal, outNorms);
}
